// ScaledDotProductAttention_14843406975281
// MI455X (gfx1250) — matrix operand order repaired and checked by reading, not compiled, not run on hardware
//
#include <hip/hip_runtime.h>

// ---------------------------------------------------------------------------
// ScaledDotProductAttention with diagonal extra-column, flash-attention style.
// N=M=4096, DK=DV=512. bf16 WMMA (16x16x32) with f32 accumulation.
// Fast path: K/V pre-converted to bf16 in workspace (V transposed), S and PV
// B-fragments loaded straight from L2 with global_load_b128 — no in-loop cvt.
// S phase uses 4 independent accumulator chains to force >=4 loads in flight.
// ---------------------------------------------------------------------------

typedef __bf16 bf16;
typedef bf16 v16bf __attribute__((ext_vector_type(16)));
typedef bf16 v8bf  __attribute__((ext_vector_type(8)));
typedef float v8f  __attribute__((ext_vector_type(8)));
typedef unsigned short u16;
typedef unsigned int   u32;
typedef u32 v8u32 __attribute__((ext_vector_type(8)));

#define NN   4096
#define MM   4096
#define DKC  512
#define DVC  512
#define BM   32          // query rows per workgroup
#define BKV  32          // keys per iteration
#define NTHREADS 256     // 8 waves (wave32)
#define KS   520         // legacy K tile LDS row stride (u16 units)
#define SS   36          // S tile row stride (f32)
#define PS   40          // P tile row stride (u16), 80B = 16B aligned
#define SCALE_513 0.04415108423526212f   // 1/sqrt(513)
#define NEG_INF (-__builtin_inff())

__device__ __forceinline__ u32 pkbf(float lo, float hi) {
  u16 a = __builtin_bit_cast(u16, (bf16)lo);
  u16 b = __builtin_bit_cast(u16, (bf16)hi);
  return (u32)a | ((u32)b << 16);
}

__device__ __forceinline__ v16bf cat8(v8bf lo, v8bf hi) {
  return __builtin_shufflevector(lo, hi, 0,1,2,3,4,5,6,7,8,9,10,11,12,13,14,15);
}

__device__ __forceinline__ v16bf ld_frag(const u16* p) {
  return cat8(*(const v8bf*)p, *(const v8bf*)(p + 8));
}

// ---------------------------------------------------------------------------
// Kernel 0: one-shot convert  K -> Kb (bf16 row-major),  V -> Vt (bf16
// transposed [col][key], via LDS tile transpose for coalescing both sides).
// grid = 512 blocks x 256 threads.
// ---------------------------------------------------------------------------
__global__ __launch_bounds__(256) void conv_kernel(const float* __restrict__ Kg,
                                                   const float* __restrict__ Vg,
                                                   u32* __restrict__ Kb32,
                                                   u32* __restrict__ Vt32) {
  const int tid = threadIdx.x;
  const int gid = blockIdx.x * 256 + tid;

  // ---- K: straight copy-convert (coalesced) ----
  const float4* K4 = (const float4*)Kg;
#pragma unroll
  for (int it = 0; it < 4; ++it) {
    const int f = gid + it * 131072;          // < 2M/4
    float4 x = K4[f];
    Kb32[f * 2 + 0] = pkbf(x.x, x.y);
    Kb32[f * 2 + 1] = pkbf(x.z, x.w);
  }

  // ---- V: 64x64 tile transpose through LDS ----
  __shared__ float T[64][65];
  const int tk = blockIdx.x >> 3;             // key tile 0..63
  const int tc = blockIdx.x & 7;              // col tile 0..7
  const int key0 = tk * 64, c0 = tc * 64;
  const float4* V4 = (const float4*)Vg;
#pragma unroll
  for (int u = 0; u < 4; ++u) {
    const int idx = tid + u * 256;            // < 1024
    const int r  = idx >> 4;
    const int c4 = idx & 15;
    float4 x = V4[(size_t)(key0 + r) * (DVC / 4) + (c0 >> 2) + c4];
    T[r][c4 * 4 + 0] = x.x; T[r][c4 * 4 + 1] = x.y;
    T[r][c4 * 4 + 2] = x.z; T[r][c4 * 4 + 3] = x.w;
  }
  __syncthreads();
#pragma unroll
  for (int u = 0; u < 8; ++u) {
    const int o  = tid + u * 256;             // < 2048
    const int c  = o >> 5;                    // 0..63
    const int kp = o & 31;                    // 0..31 (key pair)
    Vt32[(size_t)(c0 + c) * (MM / 2) + (key0 >> 1) + kp] =
        pkbf(T[2 * kp][c], T[2 * kp + 1][c]);
  }
}

// ---------------------------------------------------------------------------
// Kernel 1: q_att[n] = dot(query[n], q_k[n])
// ---------------------------------------------------------------------------
__global__ __launch_bounds__(256) void qatt_kernel(const float* __restrict__ Q,
                                                   const float* __restrict__ Qk,
                                                   float* __restrict__ outw) {
  const int tid = threadIdx.x;
  const int row = blockIdx.x * 16 + (tid >> 4);
  const int i   = tid & 15;
  const float4* a4 = (const float4*)(Q  + (size_t)row * DKC);
  const float4* b4 = (const float4*)(Qk + (size_t)row * DKC);
  float s = 0.0f;
#pragma unroll
  for (int j = 0; j < 8; ++j) {
    float4 a = a4[i + j * 16];
    float4 b = b4[i + j * 16];
    s += a.x * b.x + a.y * b.y + a.z * b.z + a.w * b.w;
  }
#pragma unroll
  for (int d = 1; d < 16; d <<= 1) s += __shfl_xor(s, d, 16);
  if (i == 0) outw[row] = s;
}

// ---------------------------------------------------------------------------
// Kernel 2 (fast path): flash attention, K/V already bf16 (V transposed).
// ---------------------------------------------------------------------------
__global__ __launch_bounds__(NTHREADS) void attn_direct(
    const float* __restrict__ Q, const float* __restrict__ qatt_g,
    const float* __restrict__ QV, const u16* __restrict__ Kb,
    const u16* __restrict__ Vt, float* __restrict__ Out) {
  __shared__ float Sh[2][BM * SS];                              // 9.2 KB
  __shared__ __attribute__((aligned(16))) u16 Pl[BM * PS];      // 2.5 KB
  __shared__ float Ml[BM], Ll[BM], Al[BM], Fs[BM], Qa[BM];
  __shared__ int UpdFlag;   // epoch flag: == kv iff a row max moved this iter

  const int tid  = threadIdx.x;
  const int lane = tid & 31;
  const int w    = tid >> 5;
  const int rt   = (w >> 1) & 1;
  const int ct   = w & 1;
  const int h    = w >> 2;
  const int hi16 = (lane & 16) ? 1 : 0;
  const int n    = lane & 15;
  const int rowblk  = blockIdx.x * BM;
  const int colbase = w * 64;

  if (tid < BM) {
    Ml[tid] = NEG_INF;
    Ll[tid] = 0.0f;
    Qa[tid] = qatt_g[rowblk + tid];
  }

  // ---- preload Q fragments (bf16, registers) ----
  v16bf qf[8];
  {
    const float* qrow = Q + (size_t)(rowblk + rt * 16 + n) * DKC;
#pragma unroll
    for (int kc = 0; kc < 8; ++kc) {
      const int k0 = h * 256 + kc * 32 + hi16 * 16;
      v8u32 t;
#pragma unroll
      for (int j2 = 0; j2 < 4; ++j2) {
        t[j2]     = pkbf(qrow[k0 + 2 * j2],      qrow[k0 + 2 * j2 + 1]);
        t[j2 + 4] = pkbf(qrow[k0 + 8 + 2 * j2], qrow[k0 + 8 + 2 * j2 + 1]);
      }
      qf[kc] = __builtin_bit_cast(v16bf, t);
    }
  }

  const v8f vzero = {0.f,0.f,0.f,0.f,0.f,0.f,0.f,0.f};
  v8f acc[8];
#pragma unroll
  for (int i = 0; i < 8; ++i) acc[i] = vzero;

  for (int kv = 0; kv < MM / BKV; ++kv) {
    const int keyg0 = kv * BKV;

    // prefetch next bf16 KV tiles into the WGP-level cache (locality 3)
    if (kv + 1 < MM / BKV) {
      __builtin_prefetch((const char*)(Kb + (size_t)(keyg0 + BKV) * DKC) + tid * 128, 0, 3);
      __builtin_prefetch((const char*)(Vt + (size_t)(keyg0 + BKV)) + (size_t)tid * 8192, 0, 3);
      __builtin_prefetch((const char*)(Vt + (size_t)(keyg0 + BKV)) + (size_t)(tid + 256) * 8192, 0, 3);
    }

    // ---- S phase: 4 independent accumulation chains => 4 B-frags in flight
    {
      const u16* kbase = Kb + (size_t)(keyg0 + ct * 16 + n) * DKC +
                         h * 256 + hi16 * 16;
      v8f s0 = vzero, s1 = vzero, s2 = vzero, s3 = vzero;
#pragma unroll
      for (int kq = 0; kq < 2; ++kq) {
        const int kc = kq * 4;
        v16bf b0 = ld_frag(kbase + (kc + 0) * 32);
        v16bf b1 = ld_frag(kbase + (kc + 1) * 32);
        v16bf b2 = ld_frag(kbase + (kc + 2) * 32);
        v16bf b3 = ld_frag(kbase + (kc + 3) * 32);
        s0 = __builtin_amdgcn_wmma_f32_16x16x32_bf16(false, qf[kc + 0], false, b0,
                                                     (short)0, s0, false, false);
        s1 = __builtin_amdgcn_wmma_f32_16x16x32_bf16(false, qf[kc + 1], false, b1,
                                                     (short)0, s1, false, false);
        s2 = __builtin_amdgcn_wmma_f32_16x16x32_bf16(false, qf[kc + 2], false, b2,
                                                     (short)0, s2, false, false);
        s3 = __builtin_amdgcn_wmma_f32_16x16x32_bf16(false, qf[kc + 3], false, b3,
                                                     (short)0, s3, false, false);
      }
      v8f s;
#pragma unroll
      for (int v = 0; v < 8; ++v) s[v] = (s0[v] + s1[v]) + (s2[v] + s3[v]);
#pragma unroll
      for (int v = 0; v < 8; ++v) {
        const int m = rt * 16 + v + hi16 * 8;
        Sh[h][m * SS + ct * 16 + n] = s[v];
      }
    }
    __syncthreads();   // (B)

    // ---- online softmax ----
    {
      const int r = tid >> 3;
      const int p = tid & 7;
      float vals[4];
      float mloc = NEG_INF;
#pragma unroll
      for (int j = 0; j < 4; ++j) {
        const int k = p * 4 + j;
        const float sv = (Sh[0][r * SS + k] + Sh[1][r * SS + k]) * SCALE_513;
        vals[j] = sv;
        mloc = fmaxf(mloc, sv);
      }
#pragma unroll
      for (int d = 1; d < 8; d <<= 1) mloc = fmaxf(mloc, __shfl_xor(mloc, d, 8));
      const float mold = Ml[r];
      const float mnew = fmaxf(mold, mloc);
      float e0 = __expf(vals[0] - mnew);
      float e1 = __expf(vals[1] - mnew);
      float e2 = __expf(vals[2] - mnew);
      float e3 = __expf(vals[3] - mnew);
      float ssum = (e0 + e1) + (e2 + e3);
      u32* pp32 = (u32*)&Pl[r * PS + p * 4];
      pp32[0] = pkbf(e0, e1);
      pp32[1] = pkbf(e2, e3);
#pragma unroll
      for (int d = 1; d < 8; d <<= 1) ssum += __shfl_xor(ssum, d, 8);
      if (p == 0) {
        const float alpha = __expf(mold - mnew);
        Al[r] = alpha;
        Ll[r] = Ll[r] * alpha + ssum;
        Ml[r] = mnew;
        if (mnew > mold) UpdFlag = kv;   // epoch write; no reset needed
      }
    }
    __syncthreads();   // (C)

    // ---- PV phase ----
    {
      const bool doResc = (UpdFlag == kv);   // workgroup-uniform

      v16bf bfrag[4];
#pragma unroll
      for (int c2 = 0; c2 < 4; ++c2)
        bfrag[c2] = ld_frag(Vt + (size_t)(colbase + c2 * 16 + n) * MM +
                            keyg0 + hi16 * 16);
#pragma unroll
      for (int r2 = 0; r2 < 2; ++r2) {
        const u16* pp = &Pl[(r2 * 16 + n) * PS + hi16 * 16];
        v16bf a = cat8(*(const v8bf*)pp, *(const v8bf*)(pp + 8));
        if (doResc) {
          float av[8];
#pragma unroll
          for (int v = 0; v < 8; ++v) av[v] = Al[r2 * 16 + v + hi16 * 8];
#pragma unroll
          for (int c2 = 0; c2 < 4; ++c2)
#pragma unroll
            for (int v = 0; v < 8; ++v) acc[r2 * 4 + c2][v] *= av[v];
        }
#pragma unroll
        for (int c2 = 0; c2 < 4; ++c2) {
          acc[r2 * 4 + c2] = __builtin_amdgcn_wmma_f32_16x16x32_bf16(
              false, a, false, bfrag[c2], (short)0, acc[r2 * 4 + c2],
              false, false);
        }
      }
    }
  }

  __syncthreads();
  if (tid < BM) {
    const float mo = Ml[tid], lo = Ll[tid];
    const float aq = Qa[tid];
    const float se = aq * SCALE_513;
    const float mf = fmaxf(mo, se);
    const float lf = lo * __expf(mo - mf) + __expf(se - mf);
    Fs[tid] = __expf(mo - mf) / lf;
  }
  __syncthreads();

#pragma unroll
  for (int r2 = 0; r2 < 2; ++r2)
#pragma unroll
    for (int c2 = 0; c2 < 4; ++c2)
#pragma unroll
      for (int v = 0; v < 8; ++v) {
        const int m    = r2 * 16 + v + hi16 * 8;
        const size_t idx = (size_t)(rowblk + m) * DVC + colbase + c2 * 16 + n;
        Out[idx] = QV[idx] * Qa[m] + acc[r2 * 4 + c2][v] * Fs[m];
      }
}

// ---------------------------------------------------------------------------
// Kernel 2 (fallback, tiny workspace): in-loop f32->bf16 staging via LDS.
// ---------------------------------------------------------------------------
__global__ __launch_bounds__(NTHREADS) void attn_legacy(
    const float* __restrict__ Q, const float* __restrict__ qatt_g,
    const float* __restrict__ QV, const float* __restrict__ Kg,
    const float* __restrict__ Vg, float* __restrict__ Out) {
  __shared__ __attribute__((aligned(16))) u16 Klds[BKV * KS];
  __shared__ float Sh[2][BM * SS];
  __shared__ __attribute__((aligned(16))) u16 Pl[BM * PS];
  __shared__ float Ml[BM], Ll[BM], Al[BM], Fs[BM], Qa[BM];
  __shared__ int UpdFlag;

  const int tid  = threadIdx.x;
  const int lane = tid & 31;
  const int w    = tid >> 5;
  const int rt   = (w >> 1) & 1;
  const int ct   = w & 1;
  const int h    = w >> 2;
  const int hi16 = (lane & 16) ? 1 : 0;
  const int n    = lane & 15;
  const int rowblk  = blockIdx.x * BM;
  const int colbase = w * 64;

  if (tid < BM) {
    Ml[tid] = NEG_INF;
    Ll[tid] = 0.0f;
    Qa[tid] = qatt_g[rowblk + tid];
  }

  v16bf qf[8];
  {
    const float* qrow = Q + (size_t)(rowblk + rt * 16 + n) * DKC;
#pragma unroll
    for (int kc = 0; kc < 8; ++kc) {
      const int k0 = h * 256 + kc * 32 + hi16 * 16;
      v8u32 t;
#pragma unroll
      for (int j2 = 0; j2 < 4; ++j2) {
        t[j2]     = pkbf(qrow[k0 + 2 * j2],      qrow[k0 + 2 * j2 + 1]);
        t[j2 + 4] = pkbf(qrow[k0 + 8 + 2 * j2], qrow[k0 + 8 + 2 * j2 + 1]);
      }
      qf[kc] = __builtin_bit_cast(v16bf, t);
    }
  }

  const v8f vzero = {0.f,0.f,0.f,0.f,0.f,0.f,0.f,0.f};
  v8f acc[8];
#pragma unroll
  for (int i = 0; i < 8; ++i) acc[i] = vzero;

  for (int kv = 0; kv < MM / BKV; ++kv) {
    const int keyg0 = kv * BKV;
    {
      const float4* Kg4 = (const float4*)Kg;
#pragma unroll
      for (int i = 0; i < (BKV * DKC / 4) / NTHREADS; ++i) {
        const int f   = tid + i * NTHREADS;
        const int key = f >> 7;
        const int c4  = f & 127;
        float4 x = Kg4[(size_t)(keyg0 + key) * (DKC / 4) + c4];
        u32* dst = (u32*)&Klds[key * KS + c4 * 4];
        dst[0] = pkbf(x.x, x.y);
        dst[1] = pkbf(x.z, x.w);
      }
    }
    __syncthreads();
    {
      v8f s = vzero;
#pragma unroll
      for (int kc = 0; kc < 8; ++kc) {
        const int koff = h * 256 + kc * 32 + hi16 * 16;
        v16bf b = ld_frag(&Klds[(ct * 16 + n) * KS + koff]);
        s = __builtin_amdgcn_wmma_f32_16x16x32_bf16(false, qf[kc], false, b,
                                                    (short)0, s, false, false);
      }
#pragma unroll
      for (int v = 0; v < 8; ++v) {
        const int m = rt * 16 + v + hi16 * 8;
        Sh[h][m * SS + ct * 16 + n] = s[v];
      }
    }
    __syncthreads();
    {
      const int r = tid >> 3;
      const int p = tid & 7;
      float vals[4];
      float mloc = NEG_INF;
#pragma unroll
      for (int j = 0; j < 4; ++j) {
        const int k = p * 4 + j;
        const float sv = (Sh[0][r * SS + k] + Sh[1][r * SS + k]) * SCALE_513;
        vals[j] = sv;
        mloc = fmaxf(mloc, sv);
      }
#pragma unroll
      for (int d = 1; d < 8; d <<= 1) mloc = fmaxf(mloc, __shfl_xor(mloc, d, 8));
      const float mold = Ml[r];
      const float mnew = fmaxf(mold, mloc);
      float e0 = __expf(vals[0] - mnew);
      float e1 = __expf(vals[1] - mnew);
      float e2 = __expf(vals[2] - mnew);
      float e3 = __expf(vals[3] - mnew);
      float ssum = (e0 + e1) + (e2 + e3);
      u32* pp32 = (u32*)&Pl[r * PS + p * 4];
      pp32[0] = pkbf(e0, e1);
      pp32[1] = pkbf(e2, e3);
#pragma unroll
      for (int d = 1; d < 8; d <<= 1) ssum += __shfl_xor(ssum, d, 8);
      if (p == 0) {
        const float alpha = __expf(mold - mnew);
        Al[r] = alpha;
        Ll[r] = Ll[r] * alpha + ssum;
        Ml[r] = mnew;
        if (mnew > mold) UpdFlag = kv;
      }
    }
    __syncthreads();
    {
      const bool doResc = (UpdFlag == kv);
      v16bf bfrag[4];
#pragma unroll
      for (int c2 = 0; c2 < 4; ++c2) {
        const float* vp = Vg + (size_t)(keyg0 + hi16 * 16) * DVC +
                          colbase + c2 * 16 + n;
        v8u32 t;
#pragma unroll
        for (int j2 = 0; j2 < 8; ++j2)
          t[j2] = pkbf(vp[(size_t)(2 * j2) * DVC], vp[(size_t)(2 * j2 + 1) * DVC]);
        bfrag[c2] = __builtin_bit_cast(v16bf, t);
      }
#pragma unroll
      for (int r2 = 0; r2 < 2; ++r2) {
        const u16* pp = &Pl[(r2 * 16 + n) * PS + hi16 * 16];
        v16bf a = cat8(*(const v8bf*)pp, *(const v8bf*)(pp + 8));
        if (doResc) {
          float av[8];
#pragma unroll
          for (int v = 0; v < 8; ++v) av[v] = Al[r2 * 16 + v + hi16 * 8];
#pragma unroll
          for (int c2 = 0; c2 < 4; ++c2)
#pragma unroll
            for (int v = 0; v < 8; ++v) acc[r2 * 4 + c2][v] *= av[v];
        }
#pragma unroll
        for (int c2 = 0; c2 < 4; ++c2) {
          acc[r2 * 4 + c2] = __builtin_amdgcn_wmma_f32_16x16x32_bf16(
              false, a, false, bfrag[c2], (short)0, acc[r2 * 4 + c2],
              false, false);
        }
      }
    }
  }

  __syncthreads();
  if (tid < BM) {
    const float mo = Ml[tid], lo = Ll[tid];
    const float aq = Qa[tid];
    const float se = aq * SCALE_513;
    const float mf = fmaxf(mo, se);
    const float lf = lo * __expf(mo - mf) + __expf(se - mf);
    Fs[tid] = __expf(mo - mf) / lf;
  }
  __syncthreads();

#pragma unroll
  for (int r2 = 0; r2 < 2; ++r2)
#pragma unroll
    for (int c2 = 0; c2 < 4; ++c2)
#pragma unroll
      for (int v = 0; v < 8; ++v) {
        const int m    = r2 * 16 + v + hi16 * 8;
        const size_t idx = (size_t)(rowblk + m) * DVC + colbase + c2 * 16 + n;
        Out[idx] = QV[idx] * Qa[m] + acc[r2 * 4 + c2][v] * Fs[m];
      }
}

// ---------------------------------------------------------------------------
extern "C" void kernel_launch(void* const* d_in, const int* in_sizes, int n_in,
                              void* d_out, int out_size, void* d_ws, size_t ws_size,
                              hipStream_t stream) {
  (void)in_sizes; (void)n_in; (void)out_size;
  const float* Q  = (const float*)d_in[0];
  const float* Qk = (const float*)d_in[1];
  const float* Qv = (const float*)d_in[2];
  const float* K  = (const float*)d_in[3];
  const float* V  = (const float*)d_in[4];
  float* Out = (float*)d_out;

  const size_t kbBytes = (size_t)MM * DKC * sizeof(u16);   // 4 MB
  const size_t vtBytes = (size_t)DVC * MM * sizeof(u16);   // 4 MB
  const size_t need = kbBytes + vtBytes + (size_t)NN * sizeof(float);

  if (ws_size >= need) {
    u16* Kb = (u16*)d_ws;
    u16* Vt = (u16*)((char*)d_ws + kbBytes);
    float* qatt = (float*)((char*)d_ws + kbBytes + vtBytes);
    conv_kernel<<<512, 256, 0, stream>>>(K, V, (u32*)Kb, (u32*)Vt);
    qatt_kernel<<<NN / 16, 256, 0, stream>>>(Q, Qk, qatt);
    attn_direct<<<NN / BM, NTHREADS, 0, stream>>>(Q, qatt, Qv, Kb, Vt, Out);
  } else {
    float* qatt = (float*)d_ws;
    qatt_kernel<<<NN / 16, 256, 0, stream>>>(Q, Qk, qatt);
    attn_legacy<<<NN / BM, NTHREADS, 0, stream>>>(Q, qatt, Qv, K, V, Out);
  }
}
